// GlimpseNetwork_4501125726782
// MI455X (gfx1250) — compile-verified, matrix operand order checked by reading, not run on hardware
//
#include <hip/hip_runtime.h>
#include <hip/hip_bf16.h>

// ---------------------------------------------------------------------------
// GlimpseNetwork for MI455X (gfx1250, wave32).
// Inputs (setup_inputs order):
//  [0] x [4096,1,100,100] f32      [1] l_t [4096,2] f32
//  [2+8t+j], t=0..2: conv_t_1_w(36), conv_t_1_b(4), bn_t_1_g(4), bn_t_1_b(4),
//                    conv_t_2_w(288), conv_t_2_b(8), bn_t_2_g(8), bn_t_2_b(8)
//  [26] fc_x_1_w(128x800) [27] b  [28] fc_x_2_w(64x800) [29] b
//  [30] fc_x_3_w(64x800)  [31] b  [32] fc_l_1_w(256x2)  [33] b
//  [34] fc_l_2_w(256x256) [35] b
// Output: g_t [4096,256] f32  ++  phi [3,4096,1,10,10] f32
// ---------------------------------------------------------------------------

#define BATCH 4096
#define HW    100

typedef __attribute__((ext_vector_type(16))) _Float16 v16h;
typedef __attribute__((ext_vector_type(8)))  _Float16 v8h;
typedef __attribute__((ext_vector_type(8)))  float    v8f;

// ---------------------------------------------------------------- utilities
__global__ void k_zero(float* __restrict__ p, int n) {
  int i = blockIdx.x * blockDim.x + threadIdx.x;
  if (i < n) p[i] = 0.f;
}

// ---------------------------------------------------------------- foveate
// phi[p][b][100]; patch sizes 10/20/40, patches 1,2 antialiased (triangle)
// downsample to 10x10.
__global__ void __launch_bounds__(128) k_foveate(
    const float* __restrict__ x, const float* __restrict__ lt,
    float* __restrict__ phi) {
  int b = blockIdx.x;
  float lx = lt[b * 2 + 0], ly = lt[b * 2 + 1];
  int sx = (int)(0.5f * (lx + 1.f) * (float)HW);  // trunc, matches .astype(i32)
  int sy = (int)(0.5f * (ly + 1.f) * (float)HW);
  const float* img = x + (size_t)b * (HW * HW);

  for (int idx = threadIdx.x; idx < 300; idx += blockDim.x) {
    int p = idx / 100, i = idx % 100;
    int oy = i / 10, ox = i % 10;
    float v = 0.f;
    if (p == 0) {
      int r = sy + oy - 5, c = sx + ox - 5;
      if (r >= 0 && r < HW && c >= 0 && c < HW) v = img[r * HW + c];
    } else if (p == 1) {
      const float w[4] = {0.125f, 0.375f, 0.375f, 0.125f};
      for (int j = 0; j < 4; ++j) {
        int ry = 2 * oy - 1 + j; ry = ry < 0 ? 0 : (ry > 19 ? 19 : ry);
        int r = sy + ry - 10;
        if (r < 0 || r >= HW) continue;
        for (int k = 0; k < 4; ++k) {
          int rx = 2 * ox - 1 + k; rx = rx < 0 ? 0 : (rx > 19 ? 19 : rx);
          int c = sx + rx - 10;
          if (c < 0 || c >= HW) continue;
          v += w[j] * w[k] * img[r * HW + c];
        }
      }
    } else {
      const float w[8] = {0.03125f, 0.09375f, 0.15625f, 0.21875f,
                          0.21875f, 0.15625f, 0.09375f, 0.03125f};
      for (int j = 0; j < 8; ++j) {
        int ry = 4 * oy - 2 + j; ry = ry < 0 ? 0 : (ry > 39 ? 39 : ry);
        int r = sy + ry - 20;
        if (r < 0 || r >= HW) continue;
        for (int k = 0; k < 8; ++k) {
          int rx = 4 * ox - 2 + k; rx = rx < 0 ? 0 : (rx > 39 ? 39 : rx);
          int c = sx + rx - 20;
          if (c < 0 || c >= HW) continue;
          v += w[j] * w[k] * img[r * HW + c];
        }
      }
    }
    phi[(size_t)p * BATCH * 100 + (size_t)b * 100 + i] = v;
  }
}

// ---------------------------------------------------------------- conv1 (1->4)
// grid (BATCH, 3); LDS-staged patch; LDS float atomics for BN stats.
__global__ void __launch_bounds__(128) k_conv1(
    const float* __restrict__ phi,
    const float* w0, const float* b0, const float* w1, const float* b1,
    const float* w2, const float* b2,
    float* __restrict__ c1, float* __restrict__ sums /*[3*4][2]*/) {
  int b = blockIdx.x, t = blockIdx.y;
  const float* cw = t == 0 ? w0 : (t == 1 ? w1 : w2);
  const float* cb = t == 0 ? b0 : (t == 1 ? b1 : b2);
  __shared__ float sp[100];
  __shared__ float ssum[4], ssq[4];
  if (threadIdx.x < 4) { ssum[threadIdx.x] = 0.f; ssq[threadIdx.x] = 0.f; }
  for (int i = threadIdx.x; i < 100; i += blockDim.x)
    sp[i] = phi[(size_t)t * BATCH * 100 + (size_t)b * 100 + i];
  __syncthreads();
  for (int idx = threadIdx.x; idx < 400; idx += blockDim.x) {
    int o = idx / 100, i = idx % 100, y = i / 10, xx = i % 10;
    float acc = cb[o];
    for (int dy = 0; dy < 3; ++dy) {
      int yy = y + dy - 1; if (yy < 0 || yy >= 10) continue;
      for (int dx = 0; dx < 3; ++dx) {
        int xc = xx + dx - 1; if (xc < 0 || xc >= 10) continue;
        acc += sp[yy * 10 + xc] * cw[o * 9 + dy * 3 + dx];
      }
    }
    c1[((size_t)t * BATCH + b) * 400 + idx] = acc;
    atomicAdd(&ssum[o], acc);
    atomicAdd(&ssq[o], acc * acc);
  }
  __syncthreads();
  if (threadIdx.x < 4) {
    atomicAdd(&sums[(t * 4 + threadIdx.x) * 2 + 0], ssum[threadIdx.x]);
    atomicAdd(&sums[(t * 4 + threadIdx.x) * 2 + 1], ssq[threadIdx.x]);
  }
}

// ---------------------------------------------------------------- BN stats
// scale = g*rstd ; shift = beta - g*rstd*mean
__global__ void k_stats(const float* __restrict__ sums,
                        const float* g0, const float* e0,
                        const float* g1, const float* e1,
                        const float* g2, const float* e2,
                        float* __restrict__ ss, int nch, float invN) {
  int i = threadIdx.x;
  if (i >= 3 * nch) return;
  int t = i / nch, c = i % nch;
  const float* g = t == 0 ? g0 : (t == 1 ? g1 : g2);
  const float* e = t == 0 ? e0 : (t == 1 ? e1 : e2);
  float m = sums[i * 2 + 0] * invN;
  float var = sums[i * 2 + 1] * invN - m * m;
  float rstd = rsqrtf(var + 1e-5f);
  float sc = g[c] * rstd;
  ss[i * 2 + 0] = sc;
  ss[i * 2 + 1] = e[c] - sc * m;
}

// ---------------------------------------------------------------- conv2 (4->8)
__global__ void __launch_bounds__(128) k_conv2(
    const float* __restrict__ c1, const float* __restrict__ ss1,
    const float* w0, const float* b0, const float* w1, const float* b1,
    const float* w2, const float* b2,
    float* __restrict__ c2, float* __restrict__ sums2 /*[3*8][2]*/) {
  int b = blockIdx.x, t = blockIdx.y;
  const float* cw = t == 0 ? w0 : (t == 1 ? w1 : w2);
  const float* cb = t == 0 ? b0 : (t == 1 ? b1 : b2);
  __shared__ float a[4][100];
  __shared__ float ssum[8], ssq[8];
  if (threadIdx.x < 8) { ssum[threadIdx.x] = 0.f; ssq[threadIdx.x] = 0.f; }
  for (int i = threadIdx.x; i < 400; i += blockDim.x) {
    int ch = i / 100, j = i % 100;
    float v = c1[((size_t)t * BATCH + b) * 400 + i];
    v = ss1[(t * 4 + ch) * 2 + 0] * v + ss1[(t * 4 + ch) * 2 + 1];
    a[ch][j] = v >= 0.f ? v : 0.01f * v;          // leaky relu
  }
  __syncthreads();
  for (int idx = threadIdx.x; idx < 800; idx += blockDim.x) {
    int o = idx / 100, i = idx % 100, y = i / 10, xx = i % 10;
    float acc = cb[o];
    for (int ic = 0; ic < 4; ++ic)
      for (int dy = 0; dy < 3; ++dy) {
        int yy = y + dy - 1; if (yy < 0 || yy >= 10) continue;
        for (int dx = 0; dx < 3; ++dx) {
          int xc = xx + dx - 1; if (xc < 0 || xc >= 10) continue;
          acc += a[ic][yy * 10 + xc] * cw[(o * 4 + ic) * 9 + dy * 3 + dx];
        }
      }
    c2[((size_t)t * BATCH + b) * 800 + idx] = acc;
    atomicAdd(&ssum[o], acc);
    atomicAdd(&ssq[o], acc * acc);
  }
  __syncthreads();
  if (threadIdx.x < 8) {
    atomicAdd(&sums2[(t * 8 + threadIdx.x) * 2 + 0], ssum[threadIdx.x]);
    atomicAdd(&sums2[(t * 8 + threadIdx.x) * 2 + 1], ssq[threadIdx.x]);
  }
}

// ---------------------------------------------------------------- features f16
__global__ void __launch_bounds__(128) k_feat(
    const float* __restrict__ c2, const float* __restrict__ ss2,
    _Float16* __restrict__ feat) {
  int b = blockIdx.x, t = blockIdx.y;
  for (int i = threadIdx.x; i < 800; i += blockDim.x) {
    int ch = i / 100;
    float v = c2[((size_t)t * BATCH + b) * 800 + i];
    v = ss2[(t * 8 + ch) * 2 + 0] * v + ss2[(t * 8 + ch) * 2 + 1];
    v = v >= 0.f ? v : 0.01f * v;
    feat[((size_t)t * BATCH + b) * 800 + i] = (_Float16)v;
  }
}

// ---------------------------------------------------------------- l-path fc1
__global__ void __launch_bounds__(256) k_lfeat(
    const float* __restrict__ lt, const float* __restrict__ w,
    const float* __restrict__ bias, _Float16* __restrict__ lf) {
  int b = blockIdx.x;
  float l0 = lt[b * 2 + 0], l1 = lt[b * 2 + 1];
  for (int n = threadIdx.x; n < 256; n += blockDim.x) {
    float v = fmaf(l0, w[n * 2 + 0], fmaf(l1, w[n * 2 + 1], bias[n]));
    lf[(size_t)b * 256 + n] = (_Float16)fmaxf(v, 0.f);
  }
}

// ---------------------------------------------------------------- weight cvt
__global__ void k_cvtw(const float* __restrict__ w1, const float* __restrict__ w2,
                       const float* __restrict__ w3, const float* __restrict__ wl2,
                       const float* __restrict__ bx1, const float* __restrict__ bx2,
                       const float* __restrict__ bx3,
                       _Float16* __restrict__ WxH, _Float16* __restrict__ Wl2H,
                       float* __restrict__ bxcat) {
  int i = blockIdx.x * blockDim.x + threadIdx.x;
  if (i < 128 * 800)      WxH[i] = (_Float16)w1[i];
  else if (i < 192 * 800) WxH[i] = (_Float16)w2[i - 128 * 800];
  else if (i < 256 * 800) WxH[i] = (_Float16)w3[i - 192 * 800];
  int j = i - 256 * 800;
  if (j >= 0 && j < 256 * 256) Wl2H[j] = (_Float16)wl2[j];
  int k = j - 256 * 256;
  if (k >= 0 && k < 256)
    bxcat[k] = k < 128 ? bx1[k] : (k < 192 ? bx2[k - 128] : bx3[k - 192]);
}

// ---------------------------------------------------------------- fused WMMA FC
// D[4096,256] = concat_t( feat_t @ Wx_t^T )  +  lf @ Wl2^T  + biases, relu.
// One wave per 16x16 tile; 25 k-steps (K=800) + 8 k-steps (K=256); f32 acc.
__device__ __forceinline__ v16h load_frag(const _Float16* p) {
  v16h f;
  *(v8h*)&f       = *(const v8h*)p;        // K halves {koff..koff+7}
  *((v8h*)&f + 1) = *(const v8h*)(p + 16); // K halves {16+koff..23+koff}
  return f;
}

__global__ void __launch_bounds__(128) k_wmma_fc(
    const _Float16* __restrict__ feat,  // [3][4096][800]
    const _Float16* __restrict__ WxH,   // [256][800] (rows: W1|W2|W3)
    const _Float16* __restrict__ lf,    // [4096][256]
    const _Float16* __restrict__ Wl2H,  // [256][256]
    const float* __restrict__ bxcat,    // [256]
    const float* __restrict__ bl2,      // [256]
    float* __restrict__ out)            // [4096][256]
{
  int wave = blockIdx.x * (blockDim.x >> 5) + (threadIdx.x >> 5);
  int lane = threadIdx.x & 31;
  int nt = wave & 15;          // 16 N-tiles (N=256)
  int mt = wave >> 4;          // 256 M-tiles (M=4096)
  int tower = (nt < 8) ? 0 : ((nt < 12) ? 1 : 2);
  int row  = lane & 15;        // A: M-row / B: N-row within tile
  int koff = (lane >> 4) * 8;  // K half-group per ISA 16-bit A layout

  const _Float16* A  = feat + ((size_t)tower * BATCH + mt * 16 + row) * 800 + koff;
  const _Float16* Bm = WxH  + (size_t)(nt * 16 + row) * 800 + koff;
  v8f c = {};
#pragma unroll 5
  for (int k0 = 0; k0 < 800; k0 += 32) {
    __builtin_prefetch((const void*)(A + k0 + 128), 0, 1);   // global_prefetch_b8
    v16h a = load_frag(A + k0);
    v16h b = load_frag(Bm + k0);
    c = __builtin_amdgcn_wmma_f32_16x16x32_f16(false, a, false, b,
                                               (short)0, c, false, false);
  }
  const _Float16* A2 = lf   + (size_t)(mt * 16 + row) * 256 + koff;
  const _Float16* B2 = Wl2H + (size_t)(nt * 16 + row) * 256 + koff;
#pragma unroll
  for (int k0 = 0; k0 < 256; k0 += 32) {
    v16h a = load_frag(A2 + k0);
    v16h b = load_frag(B2 + k0);
    c = __builtin_amdgcn_wmma_f32_16x16x32_f16(false, a, false, b,
                                               (short)0, c, false, false);
  }
  // C/D layout: lane%16 = col; VGPR r -> row (lane/16)*8 + r
  int n = nt * 16 + (lane & 15);
  float bias = bxcat[n] + bl2[n];
  int mbase = mt * 16 + (lane >> 4) * 8;
#pragma unroll
  for (int r = 0; r < 8; ++r) {
    float v = c[r] + bias;
    out[(size_t)(mbase + r) * 256 + n] = fmaxf(v, 0.f);  // relu(x_t + l)
  }
}

// ---------------------------------------------------------------------------
extern "C" void kernel_launch(void* const* d_in, const int* in_sizes, int n_in,
                              void* d_out, int out_size, void* d_ws, size_t ws_size,
                              hipStream_t stream) {
  const float* x  = (const float*)d_in[0];
  const float* lt = (const float*)d_in[1];
  auto T = [&](int t, int j) { return (const float*)d_in[2 + 8 * t + j]; };
  const float* fcx1w = (const float*)d_in[26]; const float* fcx1b = (const float*)d_in[27];
  const float* fcx2w = (const float*)d_in[28]; const float* fcx2b = (const float*)d_in[29];
  const float* fcx3w = (const float*)d_in[30]; const float* fcx3b = (const float*)d_in[31];
  const float* fcl1w = (const float*)d_in[32]; const float* fcl1b = (const float*)d_in[33];
  const float* fcl2w = (const float*)d_in[34]; const float* fcl2b = (const float*)d_in[35];

  // output layout: g_t [4096*256] ++ phi [3*4096*100]
  float* g_out   = (float*)d_out;
  float* phi_out = g_out + (size_t)BATCH * 256;

  // workspace carve-up (16B-aligned slabs)
  float*    c1    = (float*)d_ws;                      // 3*4096*400
  float*    c2    = c1 + (size_t)3 * BATCH * 400;      // 3*4096*800
  _Float16* featH = (_Float16*)(c2 + (size_t)3 * BATCH * 800); // 3*4096*800
  _Float16* lfH   = featH + (size_t)3 * BATCH * 800;   // 4096*256
  _Float16* WxH   = lfH + (size_t)BATCH * 256;         // 256*800
  _Float16* Wl2H  = WxH + 256 * 800;                   // 256*256
  float*    bxcat = (float*)(Wl2H + 256 * 256);        // 256
  float*    sums1 = bxcat + 256;                       // 24
  float*    sums2 = sums1 + 24;                        // 48
  float*    ss1   = sums2 + 48;                        // 24
  float*    ss2   = ss1 + 24;                          // 48

  // 1. zero BN accumulators
  k_zero<<<1, 128, 0, stream>>>(sums1, 72);
  // 2. foveate -> phi (final output region, also consumed by conv1)
  k_foveate<<<BATCH, 128, 0, stream>>>(x, lt, phi_out);
  // 3. fp16 weight staging + concatenated x-bias (independent of 2)
  {
    int tot = 256 * 800 + 256 * 256 + 256;
    k_cvtw<<<(tot + 255) / 256, 256, 0, stream>>>(fcx1w, fcx2w, fcx3w, fcl2w,
                                                  fcx1b, fcx2b, fcx3b,
                                                  WxH, Wl2H, bxcat);
  }
  // 4. l-path fc1 + relu -> f16
  k_lfeat<<<BATCH, 256, 0, stream>>>(lt, fcl1w, fcl1b, lfH);
  // 5. conv1 + stats pass
  k_conv1<<<dim3(BATCH, 3), 128, 0, stream>>>(
      phi_out, T(0,0), T(0,1), T(1,0), T(1,1), T(2,0), T(2,1), c1, sums1);
  k_stats<<<1, 32, 0, stream>>>(sums1, T(0,2), T(0,3), T(1,2), T(1,3),
                                T(2,2), T(2,3), ss1, 4, 1.f / (BATCH * 100.f));
  // 6. bn1+leaky+conv2 + stats pass
  k_conv2<<<dim3(BATCH, 3), 128, 0, stream>>>(
      c1, ss1, T(0,4), T(0,5), T(1,4), T(1,5), T(2,4), T(2,5), c2, sums2);
  k_stats<<<1, 32, 0, stream>>>(sums2, T(0,6), T(0,7), T(1,6), T(1,7),
                                T(2,6), T(2,7), ss2, 8, 1.f / (BATCH * 100.f));
  // 7. bn2+leaky -> f16 feature matrices [3][B][800]
  k_feat<<<dim3(BATCH, 3), 128, 0, stream>>>(c2, ss2, featH);
  // 8. fused WMMA: all fc_x GEMMs + fc_l_2 GEMM + biases + relu -> g_t
  //    4096 tiles of 16x16, 4 waves / 128-thread block
  k_wmma_fc<<<1024, 128, 0, stream>>>(featH, WxH, lfH, Wl2H, bxcat, fcl2b, g_out);
}
